// Self_Attention_65034394796806
// MI455X (gfx1250) — compile-verified, matrix operand order checked by reading, not run on hardware
//
#include <hip/hip_runtime.h>
#include <hip/hip_bf16.h>

// ---------- types ----------
typedef _Float16 half8  __attribute__((ext_vector_type(8)));
typedef _Float16 half16 __attribute__((ext_vector_type(16)));
typedef float    f8     __attribute__((ext_vector_type(8)));
typedef int      v4i    __attribute__((ext_vector_type(4)));

typedef __attribute__((address_space(1))) v4i* gptr4_t;  // global
typedef __attribute__((address_space(3))) v4i* lptr4_t;  // LDS

#define WMMA_F16F32(a, b, c) \
  __builtin_amdgcn_wmma_f32_16x16x32_f16(false, (a), false, (b), (short)0, (c), false, false)

// ---- CDNA5 async global->LDS copy (ASYNCcnt-tracked), with safe fallback ----
#if __has_builtin(__builtin_amdgcn_global_load_async_to_lds_b128)
#define ASYNC_COPY16B(gsrc, ldst)                          \
  __builtin_amdgcn_global_load_async_to_lds_b128(          \
      (gptr4_t)(void*)(gsrc), (lptr4_t)(void*)(ldst), 0, 0)
#if __has_builtin(__builtin_amdgcn_s_wait_asynccnt)
#define ASYNC_WAIT() __builtin_amdgcn_s_wait_asynccnt(0)
#else
#define ASYNC_WAIT() asm volatile("s_wait_asynccnt 0x0" ::: "memory")
#endif
#else
#define ASYNC_COPY16B(gsrc, ldst) (*(half8*)(ldst) = *(const half8*)(gsrc))
#define ASYNC_WAIT() ((void)0)
#endif

// Build a 16x16x32 f16 fragment from a row-major tile row.
// CDNA5 16-bit A/B layout: lanes 0-15 hold K {0..7,16..23},
// lanes 16-31 hold K {8..15,24..31} -> two contiguous 16B loads.
__device__ inline half16 frag32(const _Float16* __restrict__ row, int lane) {
  const int hsel = (lane >> 4) & 1;
  half8 lo = *(const half8*)(row + 8 * hsel);
  half8 hi = *(const half8*)(row + 16 + 8 * hsel);
  half16 r;
#pragma unroll
  for (int i = 0; i < 8; ++i) { r[i] = lo[i]; r[i + 8] = hi[i]; }
  return r;
}

// ---------- constants ----------
#define C_DIM 512
#define S_DIM 1024
#define B_DIM 16
#define NELT_PER_B (C_DIM * S_DIM)
#define W_ELEMS (C_DIM * C_DIM)  // 262144 per weight matrix

// ---------- kernel 0: convert 4 weight matrices f32 -> f16 (once) ----------
__global__ __launch_bounds__(256) void cvt_w_k(const float* __restrict__ w0,
                                               const float* __restrict__ w1,
                                               const float* __restrict__ w2,
                                               const float* __restrict__ w3,
                                               _Float16* __restrict__ out) {
  int idx = blockIdx.x * 256 + threadIdx.x;  // 4 * 262144
  int m = idx >> 18, r = idx & (W_ELEMS - 1);
  const float* src = (m == 0) ? w0 : (m == 1) ? w1 : (m == 2) ? w2 : w3;
  out[idx] = (_Float16)src[r];
}

// ---------- kernel 1: GroupNorm(1 group) stats per batch ----------
__global__ __launch_bounds__(256) void gn_stats_k(const float* __restrict__ x,
                                                  float* __restrict__ stats) {
  const int b = blockIdx.x;
  const float* xb = x + (size_t)b * NELT_PER_B;
  float s = 0.f, ss = 0.f;
  for (int i = threadIdx.x; i < NELT_PER_B; i += 256) {
    float v = xb[i];
    s += v; ss += v * v;
  }
  __shared__ float rs[256], rss[256];
  rs[threadIdx.x] = s; rss[threadIdx.x] = ss;
  __syncthreads();
  for (int off = 128; off > 0; off >>= 1) {
    if (threadIdx.x < off) {
      rs[threadIdx.x] += rs[threadIdx.x + off];
      rss[threadIdx.x] += rss[threadIdx.x + off];
    }
    __syncthreads();
  }
  if (threadIdx.x == 0) {
    const float inv_n = 1.0f / (float)NELT_PER_B;
    float mean = rs[0] * inv_n;
    float var = rss[0] * inv_n - mean * mean;
    stats[2 * b] = mean;
    stats[2 * b + 1] = rsqrtf(var + 1e-5f);
  }
}

// ---------- kernel 2: normalize + affine -> NT[b][s][c] f16 ----------
__global__ __launch_bounds__(256) void gn_norm_k(const float* __restrict__ x,
                                                 const float* __restrict__ gamma,
                                                 const float* __restrict__ beta,
                                                 const float* __restrict__ stats,
                                                 _Float16* __restrict__ NT) {
  size_t idx = (size_t)blockIdx.x * 256 + threadIdx.x;  // (b, s, c), c fastest
  int c = (int)(idx % C_DIM);
  size_t r = idx / C_DIM;
  int s = (int)(r % S_DIM);
  int b = (int)(r / S_DIM);
  float mean = stats[2 * b], rstd = stats[2 * b + 1];
  float v = x[((size_t)b * C_DIM + c) * S_DIM + s];
  float n = (v - mean) * rstd * gamma[c] + beta[c];
  NT[idx] = (_Float16)n;
}

// ---------- kernel 3/5: GEMM  out[o,s] = sum_c W16[o,c] * Bsrc[b][s][c] + bias[o]
// Block tile 128(o) x 128(s), 8 waves in 4x2, each wave 32x64 (2x4 WMMA tiles).
// Double-buffered LDS; tiles staged with async global->LDS copies.
// mode 0/1: write QT/KT[b][s][o]  mode 2: write V[b][o][s]
// mode 3: out32[b][o][s] = acc + bias + resid
__global__ __launch_bounds__(256, 1) void gemm512_k(const _Float16* __restrict__ Bsrc,
                                                    const _Float16* __restrict__ W16,
                                                    const float* __restrict__ bias,
                                                    const float* __restrict__ resid,
                                                    _Float16* __restrict__ out16,
                                                    float* __restrict__ out32,
                                                    int mode) {
  const int sBase = blockIdx.x * 128;
  const int oBase = blockIdx.y * 128;
  const int b = blockIdx.z;
  const int t = threadIdx.x, lane = t & 31, w = t >> 5;
  const int wm = w & 3, wn = w >> 2;  // 4 along M(o), 2 along N(s)

  __shared__ _Float16 As[2][128 * 32];
  __shared__ _Float16 Bs[2][128 * 32];

  f8 acc[2][4];
#pragma unroll
  for (int mt = 0; mt < 2; ++mt)
#pragma unroll
    for (int nt = 0; nt < 4; ++nt) acc[mt][nt] = (f8)0.0f;

  const _Float16* Bb = Bsrc + ((size_t)b * S_DIM + sBase) * C_DIM;
  const int nloc = lane & 15;
  const int srow = t >> 1, skoff = (t & 1) * 16;  // staging: 16 halves/thread

  auto stage = [&](int buf, int k0) {
    const _Float16* sa = W16 + (size_t)(oBase + srow) * C_DIM + k0 + skoff;
    const _Float16* sb = Bb + (size_t)srow * C_DIM + k0 + skoff;
    _Float16* da = &As[buf][srow * 32 + skoff];
    _Float16* db = &Bs[buf][srow * 32 + skoff];
    ASYNC_COPY16B(sa, da);
    ASYNC_COPY16B(sa + 8, da + 8);
    ASYNC_COPY16B(sb, db);
    ASYNC_COPY16B(sb + 8, db + 8);
  };

  stage(0, 0);
  ASYNC_WAIT();
  __syncthreads();

  int buf = 0;
  for (int k0 = 0; k0 < C_DIM; k0 += 32, buf ^= 1) {
    if (k0 + 32 < C_DIM) stage(buf ^ 1, k0 + 32);  // prefetch next tile

    half16 af[2], bf[4];
#pragma unroll
    for (int mt = 0; mt < 2; ++mt)
      af[mt] = frag32(&As[buf][(wm * 32 + mt * 16 + nloc) * 32], lane);
#pragma unroll
    for (int nt = 0; nt < 4; ++nt)
      bf[nt] = frag32(&Bs[buf][(wn * 64 + nt * 16 + nloc) * 32], lane);
#pragma unroll
    for (int mt = 0; mt < 2; ++mt)
#pragma unroll
      for (int nt = 0; nt < 4; ++nt)
        acc[mt][nt] = WMMA_F16F32(af[mt], bf[nt], acc[mt][nt]);

    ASYNC_WAIT();
    __syncthreads();
  }

  // epilogue: D layout — VGPR r: m = r + 8*(lane/16); n = lane&15
  const int mAdd = (lane >> 4) * 8;
#pragma unroll
  for (int mt = 0; mt < 2; ++mt) {
#pragma unroll
    for (int nt = 0; nt < 4; ++nt) {
      int sIdx = sBase + wn * 64 + nt * 16 + nloc;
#pragma unroll
      for (int r = 0; r < 8; ++r) {
        int o = oBase + wm * 32 + mt * 16 + r + mAdd;
        float v = acc[mt][nt][r] + bias[o];
        if (mode == 0 || mode == 1) {
          out16[((size_t)b * S_DIM + sIdx) * C_DIM + o] = (_Float16)v;
        } else if (mode == 2) {
          out16[((size_t)b * C_DIM + o) * S_DIM + sIdx] = (_Float16)v;
        } else {
          size_t gi = ((size_t)b * C_DIM + o) * S_DIM + sIdx;
          out32[gi] = v + resid[gi];
        }
      }
    }
  }
}

// ---------- kernel 4: fused flash attention per (b, 16-row i-block) ----------
// S = Q^T K * c^-0.5 ; online softmax over j ; O[i,c] += P V^T.
// 8 waves split head dim 512 -> 64 channels each (4 WMMA accumulators).
__global__ __launch_bounds__(256, 1) void attn_k(const _Float16* __restrict__ QT,
                                                 const _Float16* __restrict__ KT,
                                                 const _Float16* __restrict__ V,
                                                 _Float16* __restrict__ O) {
  const int i0 = blockIdx.x * 16;
  const int b = blockIdx.y;
  const int t = threadIdx.x, lane = t & 31, w = t >> 5;
  const int nloc = lane & 15, mAdd = (lane >> 4) * 8;
  const int cw = w * 64;

  __shared__ _Float16 Qs[16 * 512];      // 16KB: Q rows for this i-block
  __shared__ float Sp[8][2][256];        // per-wave partial S tiles
  __shared__ float Sred[16 * 32];        // reduced S tile
  __shared__ _Float16 Ps[16 * 32];       // softmaxed P (A operand)
  __shared__ float mrow[16], lrow[16], frow[16];

  {  // stage Qs via async copies (contiguous from QT[b][s][c])
    int row = t >> 4, col = (t & 15) * 32;
    const _Float16* src = QT + ((size_t)b * S_DIM + i0 + row) * C_DIM + col;
    _Float16* dst = &Qs[row * 512 + col];
#pragma unroll
    for (int i = 0; i < 4; ++i) ASYNC_COPY16B(src + i * 8, dst + i * 8);
  }
  if (t < 16) { mrow[t] = -1e30f; lrow[t] = 0.f; }

  f8 oacc[4];
#pragma unroll
  for (int nt = 0; nt < 4; ++nt) oacc[nt] = (f8)0.0f;
  ASYNC_WAIT();

  for (int jt = 0; jt < 32; ++jt) {
    const int j0 = jt * 32;
    __syncthreads();  // Qs ready / previous iter's Ps consumers done

    // ---- partial S over this wave's 64 channels ----
    f8 sacc[2];
    sacc[0] = (f8)0.0f; sacc[1] = (f8)0.0f;
#pragma unroll
    for (int kc = 0; kc < 2; ++kc) {
      const int kbase = cw + kc * 32;
      half16 a = frag32(&Qs[nloc * 512 + kbase], lane);
#pragma unroll
      for (int nt = 0; nt < 2; ++nt) {
        const _Float16* brow =
            KT + ((size_t)b * S_DIM + j0 + nt * 16 + nloc) * C_DIM + kbase;
        half16 bf = frag32(brow, lane);  // K rows stay hot in 192MB L2
        sacc[nt] = WMMA_F16F32(a, bf, sacc[nt]);
      }
    }
#pragma unroll
    for (int nt = 0; nt < 2; ++nt)
#pragma unroll
      for (int r = 0; r < 8; ++r) Sp[w][nt][r * 32 + lane] = sacc[nt][r];
    __syncthreads();

    // ---- cross-wave reduce into Sred[m][j] ----
    {
      const float scale = 0.044194173824159216f;  // 512^-0.5
      const int rr = t >> 5, ln = t & 31;
      const int m = rr + ((ln >> 4) * 8), n = ln & 15;
#pragma unroll
      for (int nt = 0; nt < 2; ++nt) {
        float s = 0.f;
#pragma unroll
        for (int ww = 0; ww < 8; ++ww) s += Sp[ww][nt][t];
        Sred[m * 32 + nt * 16 + n] = s * scale;
      }
    }
    __syncthreads();

    // ---- online softmax row stats + P (threads 0..15 = rows) ----
    if (t < 16) {
      float mo = mrow[t], tm = mo;
      for (int j = 0; j < 32; ++j) tm = fmaxf(tm, Sred[t * 32 + j]);
      float f = __expf(mo - tm);
      float ps = 0.f;
      for (int j = 0; j < 32; ++j) {
        float p = __expf(Sred[t * 32 + j] - tm);
        Ps[t * 32 + j] = (_Float16)p;
        ps += p;
      }
      mrow[t] = tm;
      lrow[t] = lrow[t] * f + ps;
      frow[t] = f;
    }
    __syncthreads();

    // ---- O rescale + O += P * V^T for this wave's channels ----
    half16 pa = frag32(&Ps[nloc * 32], lane);
#pragma unroll
    for (int r = 0; r < 8; ++r) {
      float f = frow[r + mAdd];
#pragma unroll
      for (int nt = 0; nt < 4; ++nt) oacc[nt][r] *= f;
    }
#pragma unroll
    for (int nt = 0; nt < 4; ++nt) {
      const _Float16* brow =
          V + ((size_t)b * C_DIM + cw + nt * 16 + nloc) * S_DIM + j0;
      half16 bf = frag32(brow, lane);
      oacc[nt] = WMMA_F16F32(pa, bf, oacc[nt]);
    }
  }

  // ---- finalize: O /= l, store O[b][i][c] f16 (coalesced across lanes) ----
#pragma unroll
  for (int nt = 0; nt < 4; ++nt) {
#pragma unroll
    for (int r = 0; r < 8; ++r) {
      int m = r + mAdd;
      float v = oacc[nt][r] / lrow[m];
      O[((size_t)b * S_DIM + i0 + m) * C_DIM + cw + nt * 16 + nloc] = (_Float16)v;
    }
  }
}

// ---------- host launch ----------
extern "C" void kernel_launch(void* const* d_in, const int* in_sizes, int n_in,
                              void* d_out, int out_size, void* d_ws, size_t ws_size,
                              hipStream_t stream) {
  const float* x     = (const float*)d_in[0];
  const float* gamma = (const float*)d_in[1];
  const float* beta  = (const float*)d_in[2];
  const float* wq    = (const float*)d_in[3];
  const float* bq    = (const float*)d_in[4];
  const float* wk    = (const float*)d_in[5];
  const float* bk    = (const float*)d_in[6];
  const float* wv    = (const float*)d_in[7];
  const float* bv    = (const float*)d_in[8];
  const float* wo    = (const float*)d_in[9];
  const float* bo    = (const float*)d_in[10];

  char* ws = (char*)d_ws;
  float* stats = (float*)ws;
  const size_t MAT = (size_t)B_DIM * S_DIM * C_DIM;  // 8.4M f16 elems = 16MB
  _Float16* W16 = (_Float16*)(ws + 4096);            // 4 x 512 x 512 f16 = 2MB
  _Float16* NT = W16 + (size_t)4 * W_ELEMS;
  _Float16* QT = NT + MAT;
  _Float16* KT = QT + MAT;
  _Float16* Vv = KT + MAT;
  _Float16* Ob = Vv + MAT;

  cvt_w_k<<<(4 * W_ELEMS) / 256, 256, 0, stream>>>(wq, wk, wv, wo, W16);
  gn_stats_k<<<B_DIM, 256, 0, stream>>>(x, stats);
  gn_norm_k<<<(B_DIM * S_DIM * C_DIM) / 256, 256, 0, stream>>>(x, gamma, beta, stats, NT);

  dim3 gGemm(S_DIM / 128, C_DIM / 128, B_DIM);
  gemm512_k<<<gGemm, 256, 0, stream>>>(NT, W16 + 0 * W_ELEMS, bq, nullptr, QT, nullptr, 0);
  gemm512_k<<<gGemm, 256, 0, stream>>>(NT, W16 + 1 * W_ELEMS, bk, nullptr, KT, nullptr, 1);
  gemm512_k<<<gGemm, 256, 0, stream>>>(NT, W16 + 2 * W_ELEMS, bv, nullptr, Vv, nullptr, 2);

  attn_k<<<dim3(S_DIM / 16, B_DIM), 256, 0, stream>>>(QT, KT, Vv, Ob);

  gemm512_k<<<gGemm, 256, 0, stream>>>(Ob, W16 + 3 * W_ELEMS, bo, x, nullptr, (float*)d_out, 3);
}
